// TAST_BN_89343909691509
// MI455X (gfx1250) — compile-verified, matrix operand order checked by reading, not run on hardware
//
#include <hip/hip_runtime.h>
#include <cstdint>

// ---------------- problem constants ----------------
#define B_ROWS 2048
#define N_SUP  16384
#define DIM    512
#define NCLS   17
#define TAU_F  10.0f
#define EPS_F  1e-12f

// ---------------- score-kernel tiling --------------
#define NSEG   8            // N split into 8 segments of 2048 for parallel top-k
#define SEGN   (N_SUP/NSEG) // 2048
#define MT     64           // z rows per workgroup
#define NCHUNK 128          // support columns per chunk
#define KSLICE 64           // K staged per LDS slice (double-buffered)
#define LDSTR  72           // padded LDS row stride (u16 elems), 16B-aligned rows
#define SCSTR  132          // padded score row stride (f32)

#define ZSL    (MT * LDSTR)       // u16 elems per z half-tile      (4608)
#define BSL    (NCHUNK * LDSTR)   // u16 elems per support half-tile (9216)
#define BUFU16 (2 * ZSL + 2 * BSL) // zh,zl,bh,bl per buffer        (27648)

typedef __bf16 v16bf __attribute__((ext_vector_type(16)));
typedef float  v8f   __attribute__((ext_vector_type(8)));
typedef int    vsi4  __attribute__((vector_size(16)));   // builtin's pointee type

// ------------- async global->LDS (gfx1250) ---------
#if defined(__has_builtin)
#  if __has_builtin(__builtin_amdgcn_global_load_async_to_lds_b128)
#    define HAVE_ASYNC_LDS 1
#  endif
#endif

__device__ __forceinline__ void cp16(const unsigned short* g, unsigned short* l) {
#ifdef HAVE_ASYNC_LDS
    // Builtin takes generic pointers to 16-byte int vectors (per clang diagnostic):
    // arg0 = global source address, arg1 = LDS destination, then imm offset, cpol.
    __builtin_amdgcn_global_load_async_to_lds_b128((vsi4*)g, (vsi4*)l, 0, 0);
#else
    *(uint4*)l = *(const uint4*)g;
#endif
}
__device__ __forceinline__ void wait_async0() {
#ifdef HAVE_ASYNC_LDS
#  if __has_builtin(__builtin_amdgcn_s_wait_asynccnt)
    __builtin_amdgcn_s_wait_asynccnt(0);
#  else
    asm volatile("s_wait_asynccnt 0x0" ::: "memory");
#  endif
#endif
}

// ---------------- helpers --------------------------
__device__ __forceinline__ unsigned short f2bf(float x) {
    unsigned int u = __float_as_uint(x);
    unsigned int r = (u + 0x7FFFu + ((u >> 16) & 1u)) >> 16;
    return (unsigned short)r;
}
__device__ __forceinline__ float bf2f(unsigned short h) {
    return __uint_as_float(((unsigned int)h) << 16);
}
__device__ __forceinline__ v8f vzero8() {
    v8f r;
#pragma unroll
    for (int i = 0; i < 8; ++i) r[i] = 0.0f;
    return r;
}
__device__ __forceinline__ v8f wmma_bf16(v16bf a, v16bf b, v8f c) {
    return __builtin_amdgcn_wmma_f32_16x16x32_bf16(false, a, false, b, (short)0, c,
                                                   false, false);
}
// Load a 16x32 bf16 WMMA fragment for this lane from LDS (row-major, LDSTR elems).
// Lane layout (ISA 7.12.2): m/n = lane&15, kbase = (lane>>4)*8;
// elements 0..7 <- K kbase..kbase+7, elements 8..15 <- K kbase+16..kbase+23.
__device__ __forceinline__ v16bf load_frag(const unsigned short* lds, int row, int kl) {
    union { v16bf v; uint4 q[2]; } f;
    const unsigned short* p = lds + row * LDSTR + kl;
    f.q[0] = *(const uint4*)(p);
    f.q[1] = *(const uint4*)(p + 16);
    return f.v;
}
__device__ __forceinline__ void topk_insert(float (&tv)[8], int (&ti)[8], float v, int n) {
    if (v <= tv[7]) return;
    tv[7] = v; ti[7] = n;
#pragma unroll
    for (int s = 7; s > 0; --s) {
        if (tv[s] > tv[s - 1]) {
            float tf = tv[s]; tv[s] = tv[s - 1]; tv[s - 1] = tf;
            int   td = ti[s]; ti[s] = ti[s - 1]; ti[s - 1] = td;
        }
    }
}

// ---------------- kernel 1: label column sums ----------------
__global__ void k_colsum(const float* __restrict__ lab, float* __restrict__ colsum) {
    const int c = blockIdx.x, t = threadIdx.x;
    __shared__ float red[256];
    float s = 0.f;
    for (int n = t; n < N_SUP; n += 256) s += lab[n * NCLS + c];
    red[t] = s; __syncthreads();
    for (int o = 128; o > 0; o >>= 1) { if (t < o) red[t] += red[t + o]; __syncthreads(); }
    if (t == 0) colsum[c] = red[0];
}

// ---------------- kernel 2: centroid partial sums (deterministic, no atomics) ----
__global__ void k_centroid(const float* __restrict__ lab, const float* __restrict__ sup,
                           float* __restrict__ part) {
    const int t = threadIdx.x;
    const int d = blockIdx.x * 256 + t;      // grid.x = 2 -> d in [0,512)
    const int seg = blockIdx.y;              // grid.y = 16 -> 1024 support rows each
    __shared__ float slab[128 * NCLS];
    float acc[NCLS];
#pragma unroll
    for (int c = 0; c < NCLS; ++c) acc[c] = 0.f;
    for (int tile = 0; tile < 8; ++tile) {
        const int n0 = seg * 1024 + tile * 128;
        __syncthreads();
        for (int i = t; i < 128 * NCLS; i += 256) slab[i] = lab[n0 * NCLS + i];
        __syncthreads();
        for (int nn = 0; nn < 128; ++nn) {
            const float sv = sup[(n0 + nn) * DIM + d];
            const float* lr = slab + nn * NCLS;
#pragma unroll
            for (int c = 0; c < NCLS; ++c) acc[c] += lr[c] * sv;
        }
    }
    float* pp = part + (size_t)seg * NCLS * DIM;
#pragma unroll
    for (int c = 0; c < NCLS; ++c) pp[c * DIM + d] = acc[c];
}

// ---------------- kernel 3: reduce partials, scale, l2-normalize centroids ------
__global__ void k_cent_norm(const float* __restrict__ part, const float* __restrict__ colsum,
                            float* __restrict__ cn) {
    const int c = blockIdx.x, t = threadIdx.x;   // 128 threads
    __shared__ float red[128];
    const float scale = 1.0f / (colsum[c] + EPS_F);
    float v[4]; float ss = 0.f;
#pragma unroll
    for (int i = 0; i < 4; ++i) {
        const int d = t * 4 + i;
        float s = 0.f;
        for (int g = 0; g < 16; ++g) s += part[(size_t)g * NCLS * DIM + c * DIM + d];
        v[i] = s * scale;
        ss += v[i] * v[i];
    }
    red[t] = ss; __syncthreads();
    for (int o = 64; o > 0; o >>= 1) { if (t < o) red[t] += red[t + o]; __syncthreads(); }
    const float inv = 1.0f / fmaxf(sqrtf(red[0]), EPS_F);
#pragma unroll
    for (int i = 0; i < 4; ++i) cn[c * DIM + t * 4 + i] = v[i] * inv;
}

// ---------------- kernel 4: normalize z rows, emit bf16 hi/lo + query logits ----
__global__ void k_z(const float* __restrict__ z, const float* __restrict__ cn,
                    unsigned short* __restrict__ zhi, unsigned short* __restrict__ zlo,
                    float* __restrict__ logits) {
    const int b = blockIdx.x, t = threadIdx.x;   // 128 threads
    __shared__ float red[128];
    __shared__ float sdot[NCLS * 128];
    const float4 v4 = ((const float4*)(z + (size_t)b * DIM))[t];
    float vv[4] = {v4.x, v4.y, v4.z, v4.w};
    red[t] = vv[0]*vv[0] + vv[1]*vv[1] + vv[2]*vv[2] + vv[3]*vv[3];
    __syncthreads();
    for (int o = 64; o > 0; o >>= 1) { if (t < o) red[t] += red[t + o]; __syncthreads(); }
    const float inv = 1.0f / fmaxf(sqrtf(red[0]), EPS_F);
    float nr[4]; unsigned short hs[4], ls[4];
#pragma unroll
    for (int i = 0; i < 4; ++i) {
        nr[i] = vv[i] * inv;
        hs[i] = f2bf(nr[i]);
        ls[i] = f2bf(nr[i] - bf2f(hs[i]));
    }
    ushort4 h4; h4.x = hs[0]; h4.y = hs[1]; h4.z = hs[2]; h4.w = hs[3];
    ushort4 l4; l4.x = ls[0]; l4.y = ls[1]; l4.z = ls[2]; l4.w = ls[3];
    *(ushort4*)(zhi + (size_t)b * DIM + t * 4) = h4;
    *(ushort4*)(zlo + (size_t)b * DIM + t * 4) = l4;
#pragma unroll
    for (int c = 0; c < NCLS; ++c) {
        const float* cr = cn + c * DIM + t * 4;
        sdot[c * 128 + t] = nr[0]*cr[0] + nr[1]*cr[1] + nr[2]*cr[2] + nr[3]*cr[3];
    }
    __syncthreads();
    if (t < NCLS) {
        float s = 0.f;
        for (int i = 0; i < 128; ++i) s += sdot[t * 128 + i];
        logits[b * NCLS + t] = TAU_F * s;
    }
}

// ---------------- kernel 5: normalize supports, YY, support logits -> softmax/argmax
__global__ void k_supports(const float* __restrict__ sup, const float* __restrict__ cn,
                           unsigned short* __restrict__ shi, unsigned short* __restrict__ slo,
                           float* __restrict__ yy, float* __restrict__ tlsm,
                           int* __restrict__ tlarg) {
    const int n = blockIdx.x, t = threadIdx.x;   // 128 threads
    __shared__ float red[128];
    __shared__ float sdot[NCLS * 128];
    __shared__ float stl[NCLS];
    const float4 v4 = ((const float4*)(sup + (size_t)n * DIM))[t];
    float vv[4] = {v4.x, v4.y, v4.z, v4.w};
    red[t] = vv[0]*vv[0] + vv[1]*vv[1] + vv[2]*vv[2] + vv[3]*vv[3];
    __syncthreads();
    for (int o = 64; o > 0; o >>= 1) { if (t < o) red[t] += red[t + o]; __syncthreads(); }
    const float inv = 1.0f / fmaxf(sqrtf(red[0]), EPS_F);
    float nr[4]; unsigned short hs[4], ls[4];
    float ysq = 0.f;
#pragma unroll
    for (int i = 0; i < 4; ++i) {
        nr[i] = vv[i] * inv;
        ysq += nr[i] * nr[i];
        hs[i] = f2bf(nr[i]);
        ls[i] = f2bf(nr[i] - bf2f(hs[i]));
    }
    ushort4 h4; h4.x = hs[0]; h4.y = hs[1]; h4.z = hs[2]; h4.w = hs[3];
    ushort4 l4; l4.x = ls[0]; l4.y = ls[1]; l4.z = ls[2]; l4.w = ls[3];
    *(ushort4*)(shi + (size_t)n * DIM + t * 4) = h4;
    *(ushort4*)(slo + (size_t)n * DIM + t * 4) = l4;
    __syncthreads();                // protect red[] reuse
    red[t] = ysq; __syncthreads();
    for (int o = 64; o > 0; o >>= 1) { if (t < o) red[t] += red[t + o]; __syncthreads(); }
    if (t == 0) yy[n] = red[0];
#pragma unroll
    for (int c = 0; c < NCLS; ++c) {
        const float* cr = cn + c * DIM + t * 4;
        sdot[c * 128 + t] = nr[0]*cr[0] + nr[1]*cr[1] + nr[2]*cr[2] + nr[3]*cr[3];
    }
    __syncthreads();
    if (t < NCLS) {
        float s = 0.f;
        for (int i = 0; i < 128; ++i) s += sdot[t * 128 + i];
        stl[t] = TAU_F * s;
    }
    __syncthreads();
    if (t == 0) {
        float m = stl[0]; int am = 0;
        for (int c = 1; c < NCLS; ++c) if (stl[c] > m) { m = stl[c]; am = c; }
        float e[NCLS], se = 0.f;
        for (int c = 0; c < NCLS; ++c) { e[c] = expf(stl[c] - m); se += e[c]; }
        for (int c = 0; c < NCLS; ++c) tlsm[(size_t)n * NCLS + c] = e[c] / se;
        tlarg[n] = am;
    }
}

// ---------------- kernel 6: fused bf16x3 WMMA similarity + streaming top-8 -------
// Double-buffered K-slices; async global->LDS staging overlapped with WMMA.
__global__ __launch_bounds__(256)
void k_scores(const unsigned short* __restrict__ zhi_g, const unsigned short* __restrict__ zlo_g,
              const unsigned short* __restrict__ shi_g, const unsigned short* __restrict__ slo_g,
              const float* __restrict__ yy,
              float* __restrict__ candv, int* __restrict__ candi) {
    extern __shared__ unsigned char smem[];
    unsigned short* base = (unsigned short*)smem;            // 2 x BUFU16
    float* scores = (float*)(base + 2 * BUFU16);             // MT x SCSTR
    float* cv = scores;                                      // overlay: used after last scan
    int*   ci = (int*)(scores + MT * 32);

    const int tid  = threadIdx.x;
    const int mt   = blockIdx.x;          // 0..31 (M tiles of 64)
    const int seg  = blockIdx.y;          // 0..7
    const int segBase = seg * SEGN;

    const int w    = tid >> 5;            // wave 0..7
    const int lane = tid & 31;
    const int mb   = w & 3;               // wave's 16-row block (0..3)
    const int h    = w >> 2;              // wave's 64-col half  (0..1)
    const int nl   = lane & 15;
    const int kb8  = (lane >> 4) * 8;     // lane K sub-offset

    const int srow = tid & 63, sq = tid >> 6;   // scan mapping: 4 threads/row

    float tv[8]; int ti[8];
#pragma unroll
    for (int j = 0; j < 8; ++j) { tv[j] = -1e30f; ti[j] = 0; }

    const int NCH = SEGN / NCHUNK;        // 16 chunks
    const int SPC = DIM / KSLICE;         // 8 K-slices per chunk
    const int TOT = NCH * SPC;            // 128 slices total

    // stage one K-slice (slice index u decomposed into chunk + K offset)
    auto stage = [&](int u) {
        unsigned short* buf = base + (u & 1) * BUFU16;
        unsigned short* zh = buf;
        unsigned short* zl = buf + ZSL;
        unsigned short* bh = buf + 2 * ZSL;
        unsigned short* bl = buf + 2 * ZSL + BSL;
        const int nB = segBase + (u / SPC) * NCHUNK;
        const int s  = u % SPC;
        for (int i = tid; i < MT * 8; i += 256) {
            const int r = i >> 3, c8 = (i & 7) * 8;
            const size_t g = (size_t)(mt * MT + r) * DIM + s * KSLICE + c8;
            cp16(zhi_g + g, zh + r * LDSTR + c8);
            cp16(zlo_g + g, zl + r * LDSTR + c8);
        }
        for (int i = tid; i < NCHUNK * 8; i += 256) {
            const int r = i >> 3, c8 = (i & 7) * 8;
            const size_t g = (size_t)(nB + r) * DIM + s * KSLICE + c8;
            cp16(shi_g + g, bh + r * LDSTR + c8);
            cp16(slo_g + g, bl + r * LDSTR + c8);
        }
    };

    stage(0);   // preload first slice

    for (int chunk = 0; chunk < NCH; ++chunk) {
        const int nBase = segBase + chunk * NCHUNK;
        v8f acc[4];
#pragma unroll
        for (int nb = 0; nb < 4; ++nb) acc[nb] = vzero8();

        for (int s = 0; s < SPC; ++s) {
            const int u = chunk * SPC + s;
            wait_async0();        // slice u resident (own async loads done)
            __syncthreads();      // all waves' staging of slice u visible
            if (u + 1 < TOT) stage(u + 1);   // in flight under the WMMAs below

            const unsigned short* buf = base + (u & 1) * BUFU16;
            const unsigned short* zh = buf;
            const unsigned short* zl = buf + ZSL;
            const unsigned short* bh = buf + 2 * ZSL;
            const unsigned short* bl = buf + 2 * ZSL + BSL;
#pragma unroll
            for (int j = 0; j < KSLICE / 32; ++j) {
                const int kl = j * 32 + kb8;
                v16bf ah = load_frag(zh, mb * 16 + nl, kl);
                v16bf al = load_frag(zl, mb * 16 + nl, kl);
#pragma unroll
                for (int nb = 0; nb < 4; ++nb) {
                    const int brow = h * 64 + nb * 16 + nl;
                    v16bf bhf = load_frag(bh, brow, kl);
                    v16bf blf = load_frag(bl, brow, kl);
                    acc[nb] = wmma_bf16(ah, bhf, acc[nb]);   // hi*hi
                    acc[nb] = wmma_bf16(ah, blf, acc[nb]);   // hi*lo
                    acc[nb] = wmma_bf16(al, bhf, acc[nb]);   // lo*hi
                }
            }
        }
        // write selection keys: key = 2*dot - YY[n]  (monotone in exp(-dist))
#pragma unroll
        for (int nb = 0; nb < 4; ++nb) {
            const int col = h * 64 + nb * 16 + nl;
            const float yv = yy[nBase + col];
#pragma unroll
            for (int r = 0; r < 8; ++r) {
                const int ml = (lane < 16) ? r : (r + 8);
                scores[(mb * 16 + ml) * SCSTR + col] = 2.0f * acc[nb][r] - yv;
            }
        }
        __syncthreads();
        // each thread scans its 32 columns of its row
        const int cbase = sq * 32;
        for (int c = 0; c < 32; ++c) {
            topk_insert(tv, ti, scores[srow * SCSTR + cbase + c], nBase + cbase + c);
        }
        __syncthreads();
    }
    // merge 4 per-row lists -> 8 candidates per (row, seg); score tile is dead now
#pragma unroll
    for (int j = 0; j < 8; ++j) { cv[srow * 32 + sq * 8 + j] = tv[j]; ci[srow * 32 + sq * 8 + j] = ti[j]; }
    __syncthreads();
    if (tid < MT) {
        float mv[8]; int mi[8];
#pragma unroll
        for (int j = 0; j < 8; ++j) { mv[j] = -1e30f; mi[j] = 0; }
        for (int j = 0; j < 32; ++j) topk_insert(mv, mi, cv[tid * 32 + j], ci[tid * 32 + j]);
        const size_t obase = (size_t)(mt * MT + tid) * (NSEG * 8) + seg * 8;
#pragma unroll
        for (int j = 0; j < 8; ++j) { candv[obase + j] = mv[j]; candi[obase + j] = mi[j]; }
    }
}

// ---------------- kernel 7: merge segments, gather targets/outputs ---------------
__global__ void k_merge(const float* __restrict__ candv, const int* __restrict__ candi,
                        const int* __restrict__ tlarg, const float* __restrict__ tlsm,
                        float* __restrict__ out_t, float* __restrict__ out_o) {
    const int b = blockIdx.x, t = threadIdx.x;   // 32 threads
    __shared__ int sidx[8];
    __shared__ float st[NCLS], so[NCLS], ssum[2];
    if (t == 0) {
        float tv[8]; int ti[8];
#pragma unroll
        for (int j = 0; j < 8; ++j) { tv[j] = -1e30f; ti[j] = 0; }
        for (int j = 0; j < NSEG * 8; ++j)
            topk_insert(tv, ti, candv[(size_t)b * NSEG * 8 + j], candi[(size_t)b * NSEG * 8 + j]);
#pragma unroll
        for (int j = 0; j < 8; ++j) sidx[j] = ti[j];
    }
    __syncthreads();
    if (t < NCLS) {
        float ts = 0.f, os = 0.f;
#pragma unroll
        for (int j = 0; j < 8; ++j) {
            const int n = sidx[j];
            ts += (tlarg[n] == t) ? 1.0f : 0.0f;
            os += tlsm[(size_t)n * NCLS + t];
        }
        st[t] = ts; so[t] = os;
    }
    __syncthreads();
    if (t == 0) {
        float a = 0.f, bb = 0.f;
        for (int c = 0; c < NCLS; ++c) { a += st[c]; bb += so[c]; }
        ssum[0] = a; ssum[1] = bb;
    }
    __syncthreads();
    if (t < NCLS) {
        out_t[b * NCLS + t] = st[t] / (ssum[0] + EPS_F);
        out_o[b * NCLS + t] = so[t] / (ssum[1] + EPS_F);
    }
}

// ---------------- launch ---------------------------------------------------------
extern "C" void kernel_launch(void* const* d_in, const int* in_sizes, int n_in,
                              void* d_out, int out_size, void* d_ws, size_t ws_size,
                              hipStream_t stream) {
    (void)in_sizes; (void)n_in; (void)out_size; (void)ws_size;
    const float* z   = (const float*)d_in[0];
    const float* sup = (const float*)d_in[1];
    const float* lab = (const float*)d_in[2];
    float* out         = (float*)d_out;
    float* out_logits  = out;
    float* out_targets = out + (size_t)B_ROWS * NCLS;
    float* out_outputs = out + (size_t)2 * B_ROWS * NCLS;

    uintptr_t p = (uintptr_t)d_ws;
    auto alloc = [&](size_t bytes) -> void* {
        void* r = (void*)p;
        p += (bytes + 255) & ~(size_t)255;
        return r;
    };
    float* colsum = (float*)alloc(NCLS * sizeof(float));
    float* part   = (float*)alloc((size_t)16 * NCLS * DIM * sizeof(float));
    float* cn     = (float*)alloc((size_t)NCLS * DIM * sizeof(float));
    float* yy     = (float*)alloc((size_t)N_SUP * sizeof(float));
    float* tlsm   = (float*)alloc((size_t)N_SUP * NCLS * sizeof(float));
    int*   tlarg  = (int*)alloc((size_t)N_SUP * sizeof(int));
    float* candv  = (float*)alloc((size_t)B_ROWS * NSEG * 8 * sizeof(float));
    int*   candi  = (int*)alloc((size_t)B_ROWS * NSEG * 8 * sizeof(int));
    unsigned short* zhi = (unsigned short*)alloc((size_t)B_ROWS * DIM * 2);
    unsigned short* zlo = (unsigned short*)alloc((size_t)B_ROWS * DIM * 2);
    unsigned short* shi = (unsigned short*)alloc((size_t)N_SUP * DIM * 2);
    unsigned short* slo = (unsigned short*)alloc((size_t)N_SUP * DIM * 2);

    k_colsum<<<NCLS, 256, 0, stream>>>(lab, colsum);
    k_centroid<<<dim3(2, 16), 256, 0, stream>>>(lab, sup, part);
    k_cent_norm<<<NCLS, 128, 0, stream>>>(part, colsum, cn);
    k_z<<<B_ROWS, 128, 0, stream>>>(z, cn, zhi, zlo, out_logits);
    k_supports<<<N_SUP, 128, 0, stream>>>(sup, cn, shi, slo, yy, tlsm, tlarg);

    const size_t smem = (size_t)2 * BUFU16 * sizeof(unsigned short) +
                        (size_t)MT * SCSTR * sizeof(float);
    k_scores<<<dim3(B_ROWS / MT, NSEG), 256, smem, stream>>>(zhi, zlo, shi, slo, yy,
                                                             candv, candi);
    k_merge<<<B_ROWS, 32, 0, stream>>>(candv, candi, tlarg, tlsm, out_targets, out_outputs);
}
